// PairEnergies_35708358099590
// MI455X (gfx1250) — matrix operand order repaired and checked by reading, not compiled, not run on hardware
//
#include <hip/hip_runtime.h>
#include <hip/hip_bf16.h>

// Problem constants (match reference)
constexpr int cB = 4, cN = 1024, cK = 30, cH = 128, cOUT = 400, cFF = 512;
constexpr int NEDGE = cB * cN * cK;   // 122880 (divisible by 16)
constexpr int NNODE = cB * cN;        // 4096

typedef __attribute__((ext_vector_type(16))) _Float16 v16h;
typedef __attribute__((ext_vector_type(8)))  float    v8f;

// ---------------------------------------------------------------------------
// WMMA helpers (gfx1250 wave32, 16x16x32 f16 -> f32)
// ---------------------------------------------------------------------------
__device__ __forceinline__ v8f wmma_f16(v16h a, v16h b, v8f c) {
  return __builtin_amdgcn_wmma_f32_16x16x32_f16(false, a, false, b, (short)0, c,
                                                false, false);
}

// A-matrix fragment 16x32 f16 from LDS (row-major, row stride lda elements).
// ISA 7.12.2: lanes 0-15 hold row M=lane, K 0..7 / 16..23 ;
// lanes 16-31 hold row M=lane-16, K 8..15 / 24..31.  Each 8-element run is
// contiguous in K -> compiler emits 2x ds_load_b128 per fragment.
__device__ __forceinline__ v16h load_a_frag_lds(const _Float16* A, int lda, int kbase) {
  int l   = threadIdx.x & 31;
  int row = l & 15;
  int khi = (l >> 4) << 3; // 0 or 8
  v16h a;
#pragma unroll
  for (int e = 0; e < 16; ++e) {
    int k = kbase + khi + ((e < 8) ? e : (e + 8));
    a[e] = A[row * lda + k];
  }
  return a;
}

// B-matrix fragment from pre-packed weights: fragment-major layout, one
// contiguous 32-byte load per lane (global_load_b128 x2, coalesced).
__device__ __forceinline__ v16h load_b_packed(const _Float16* Wp, int KT, int kt, int nt) {
  int l = threadIdx.x & 31;
  return *reinterpret_cast<const v16h*>(Wp + ((((size_t)nt * KT + kt) * 32 + l) << 4));
}

// LayerNorm over a 16x128 fp32 tile in LDS; 256 threads. rs/rq are 256-float LDS.
__device__ __forceinline__ void ln_tile_16x128(float* T, const float* g, const float* b,
                                               float* rs, float* rq) {
  int tid = threadIdx.x;
  int row = tid >> 4, sub = tid & 15;
  float s = 0.f, q = 0.f;
#pragma unroll
  for (int e = 0; e < 8; ++e) { float x = T[row * 128 + sub * 8 + e]; s += x; q += x * x; }
  rs[tid] = s; rq[tid] = q;
  __syncthreads();
  if (sub == 0) {
    float ss = 0.f, qq = 0.f;
#pragma unroll
    for (int e = 0; e < 16; ++e) { ss += rs[(row << 4) + e]; qq += rq[(row << 4) + e]; }
    float mean = ss * (1.f / 128.f);
    float var  = qq * (1.f / 128.f) - mean * mean;
    rs[row << 4] = mean;
    rq[row << 4] = rsqrtf(var + 1e-5f);
  }
  __syncthreads();
  float mean = rs[row << 4], inv = rq[row << 4];
#pragma unroll
  for (int e = 0; e < 8; ++e) {
    int c = sub * 8 + e;
    float x = T[row * 128 + c];
    T[row * 128 + c] = (x - mean) * inv * g[c] + b[c];
  }
  __syncthreads();
}

__device__ __forceinline__ float block_sum_128(float v, float* red) {
  int c = threadIdx.x;
  red[c] = v; __syncthreads();
  for (int s = 64; s > 0; s >>= 1) { if (c < s) red[c] += red[c + s]; __syncthreads(); }
  float r = red[0]; __syncthreads();
  return r;
}

// ---------------------------------------------------------------------------
// Feature kernels
// ---------------------------------------------------------------------------
__device__ __forceinline__ void load_xb(const float* X, int b, int s, float o[3]) {
  const float* p = X + (((size_t)b * cN + (s / 3)) * 4 + (s % 3)) * 3;
  o[0] = p[0]; o[1] = p[1]; o[2] = p[2];
}
__device__ __forceinline__ void unit_u(const float* X, int b, int s, float u[3]) {
  float p0[3], p1[3]; load_xb(X, b, s, p0); load_xb(X, b, s + 1, p1);
  float d0 = p1[0] - p0[0], d1 = p1[1] - p0[1], d2 = p1[2] - p0[2];
  float nrm = fmaxf(sqrtf(d0 * d0 + d1 * d1 + d2 * d2), 1e-7f);
  u[0] = d0 / nrm; u[1] = d1 / nrm; u[2] = d2 / nrm;
}

__global__ void dihedral_kernel(const float* __restrict__ X, float* __restrict__ Vf) {
  int g = blockIdx.x * blockDim.x + threadIdx.x;
  if (g >= cB * cN * 3) return;
  int b = g / (cN * 3); int rem = g - b * (cN * 3);
  int n = rem / 3, d = rem - n * 3;
  int t = n * 3 + d;
  float D = 0.f;
  if (t >= 1 && t < 3 * cN - 2) {
    int tt = t - 1;
    float u2[3], u1[3], u0[3];
    unit_u(X, b, tt, u2); unit_u(X, b, tt + 1, u1); unit_u(X, b, tt + 2, u0);
    float n2[3] = { u2[1]*u1[2]-u2[2]*u1[1], u2[2]*u1[0]-u2[0]*u1[2], u2[0]*u1[1]-u2[1]*u1[0] };
    float n1[3] = { u1[1]*u0[2]-u1[2]*u0[1], u1[2]*u0[0]-u1[0]*u0[2], u1[0]*u0[1]-u1[1]*u0[0] };
    float l2 = fmaxf(sqrtf(n2[0]*n2[0]+n2[1]*n2[1]+n2[2]*n2[2]), 1e-7f);
    float l1 = fmaxf(sqrtf(n1[0]*n1[0]+n1[1]*n1[1]+n1[2]*n1[2]), 1e-7f);
    float cd = (n2[0]*n1[0]+n2[1]*n1[1]+n2[2]*n1[2]) / (l2 * l1);
    cd = fminf(fmaxf(cd, -1.f + 1e-7f), 1.f - 1e-7f);
    float sg = u2[0]*n1[0] + u2[1]*n1[1] + u2[2]*n1[2];
    float sgn = (sg > 0.f) ? 1.f : ((sg < 0.f) ? -1.f : 0.f);
    D = sgn * acosf(cd);
  }
  Vf[((size_t)b * cN + n) * 6 + d]     = cosf(D);
  Vf[((size_t)b * cN + n) * 6 + 3 + d] = sinf(D);
}

// One block per (b,i): distances to all N nodes in LDS, 30x argmin with
// lowest-index tie-break (matches jax.lax.top_k on negated distances).
__global__ void topk_kernel(const float* __restrict__ X, const float* __restrict__ xmask,
                            int* __restrict__ Eidx, float* __restrict__ Dn) {
  __shared__ float dist[cN];
  __shared__ float rv[256];
  __shared__ int   ri[256];
  int tid = threadIdx.x;
  int b = blockIdx.x / cN, i = blockIdx.x - b * cN;
  const float* ca_i = X + (((size_t)b * cN + i) * 4 + 1) * 3;
  float xi = ca_i[0], yi = ca_i[1], zi = ca_i[2];
  float mi = xmask[b * cN + i];
  for (int n = tid; n < cN; n += 256) {
    const float* ca = X + (((size_t)b * cN + n) * 4 + 1) * 3;
    float dx = xi - ca[0], dy = yi - ca[1], dz = zi - ca[2];
    float d = sqrtf(dx * dx + dy * dy + dz * dz + 1e-8f);
    float mm = mi * xmask[b * cN + n];
    dist[n] = (mm > 0.f) ? d : 1e6f;
  }
  __syncthreads();
  for (int k = 0; k < cK; ++k) {
    float bv = 3.3e38f; int bi = cN;
    for (int n = tid; n < cN; n += 256) {
      float v = dist[n];
      if (v < bv) { bv = v; bi = n; }
    }
    rv[tid] = bv; ri[tid] = bi;
    __syncthreads();
    for (int s = 128; s > 0; s >>= 1) {
      if (tid < s) {
        if (rv[tid + s] < rv[tid] ||
            (rv[tid + s] == rv[tid] && ri[tid + s] < ri[tid])) {
          rv[tid] = rv[tid + s]; ri[tid] = ri[tid + s];
        }
      }
      __syncthreads();
    }
    if (tid == 0) {
      Eidx[(size_t)blockIdx.x * cK + k] = ri[0];
      Dn  [(size_t)blockIdx.x * cK + k] = rv[0];
      dist[ri[0]] = 3.4e38f;
    }
    __syncthreads();
  }
}

// Edge features: 16 RBF + 16 sinusoidal positional -> f16, 32 per edge.
__global__ void ef_kernel(const float* __restrict__ Dn, const int* __restrict__ Eidx,
                          _Float16* __restrict__ Ef16) {
  int r = blockIdx.x * blockDim.x + threadIdx.x;
  if (r >= NEDGE) return;
  int b = r / (cN * cK); int rem = r - b * (cN * cK);
  int i = rem / cK;
  float Dv  = Dn[r];
  float off = (float)(Eidx[r] - i);
  _Float16* o = Ef16 + (size_t)r * 32;
#pragma unroll
  for (int m = 0; m < 16; ++m) {
    float ctr = 2.0f + (20.0f / 15.0f) * (float)m;
    float z = (Dv - ctr) * (1.0f / 1.25f);
    o[m] = (_Float16)expf(-z * z);
  }
#pragma unroll
  for (int p = 0; p < 8; ++p) {
    float f = expf(-1.1512925464970229f * (float)p); // ln(1e4)/8
    float ang = off * f;
    o[16 + p] = (_Float16)cosf(ang);
    o[24 + p] = (_Float16)sinf(ang);
  }
}

// ---------------------------------------------------------------------------
// Embedding kernels
// ---------------------------------------------------------------------------
__global__ void v_embed_kernel(const float* __restrict__ Vf,
                               const float* fVw, const float* fVb,
                               const float* fVg, const float* fVbeta,
                               const float* Wvw, const float* Wvb,
                               float* __restrict__ hV) {
  __shared__ float sV[128];
  __shared__ float red[128];
  int node = blockIdx.x, c = threadIdx.x;
  float a = fVb[c];
#pragma unroll
  for (int k = 0; k < 6; ++k) a += Vf[(size_t)node * 6 + k] * fVw[k * 128 + c];
  float mean = block_sum_128(a, red) * (1.f / 128.f);
  float d = a - mean;
  float var = block_sum_128(d * d, red) * (1.f / 128.f);
  sV[c] = d * rsqrtf(var + 1e-5f) * fVg[c] + fVbeta[c];
  __syncthreads();
  float h = Wvb[c];
  for (int k = 0; k < 128; ++k) h += sV[k] * Wvw[k * 128 + c];
  hV[(size_t)node * 128 + c] = h;
}

// E-embed: hE = LN(Ef @ fEw + fEb) @ Wew + Web, fused per 16-row tile.
__global__ void e_embed_kernel(const _Float16* __restrict__ Ef16,
                               const _Float16* __restrict__ fEwP, const float* fEb,
                               const float* fEg, const float* fEbeta,
                               const _Float16* __restrict__ WewP, const float* Web,
                               float* __restrict__ hE) {
  __shared__ _Float16 sA[16 * 32];
  __shared__ float    sT[16 * 128];
  __shared__ _Float16 sE[16 * 128];
  __shared__ float rs[256], rq[256];
  int tid = threadIdx.x;
  int r0 = blockIdx.x * 16;
  for (int idx = tid; idx < 16 * 32; idx += 256) sA[idx] = Ef16[(size_t)r0 * 32 + idx];
  __syncthreads();
  int w = tid >> 5, l = tid & 31;
  int n = (l & 15) + w * 16, mb = (l >> 4) ? 8 : 0;
  v8f acc = {};
  {
    v16h a  = load_a_frag_lds(sA, 32, 0);
    v16h bf = load_b_packed(fEwP, 1, 0, w);
    acc = wmma_f16(a, bf, acc);
  }
#pragma unroll
  for (int e = 0; e < 8; ++e) sT[(mb + e) * 128 + n] = acc[e] + fEb[n];
  __syncthreads();
  ln_tile_16x128(sT, fEg, fEbeta, rs, rq);
  for (int idx = tid; idx < 16 * 128; idx += 256) sE[idx] = (_Float16)sT[idx];
  __syncthreads();
  v8f acc2 = {};
#pragma unroll
  for (int kt = 0; kt < 4; ++kt) {
    v16h a  = load_a_frag_lds(sE, 128, kt * 32);
    v16h bf = load_b_packed(WewP, 4, kt, w);
    acc2 = wmma_f16(a, bf, acc2);
  }
#pragma unroll
  for (int e = 0; e < 8; ++e) hE[(size_t)(r0 + mb + e) * 128 + n] = acc2[e] + Web[n];
}

// ---------------------------------------------------------------------------
// Fused 3-GEMM message chain. mode 0: edge ([hE,hVi,hVj] -> residual+LN into hE)
//                             mode 1: node ([hVi,hVj,hE] -> masked mn_out)
// ---------------------------------------------------------------------------
__global__ void msg_kernel(const float* __restrict__ hV, float* __restrict__ hE,
                           float* __restrict__ mn_out,
                           const int* __restrict__ Eidx, const float* __restrict__ xmask,
                           const _Float16* __restrict__ W1, const float* __restrict__ b1,
                           const _Float16* __restrict__ W2, const float* __restrict__ b2,
                           const _Float16* __restrict__ W3, const float* __restrict__ b3,
                           const float* __restrict__ Lg, const float* __restrict__ Lb,
                           int mode) {
  __shared__ _Float16 sA[16 * 384];
  __shared__ _Float16 sM[16 * 128];
  __shared__ float    sT[16 * 128];
  __shared__ float rs[256], rq[256];
  __shared__ int   sI[16], sJ[16];
  __shared__ float sMask[16];
  int tid = threadIdx.x;
  int r0 = blockIdx.x * 16;
  if (tid < 16) {
    int r = r0 + tid;
    int b = r / (cN * cK);
    int rem = r - b * (cN * cK);
    int i = rem / cK;
    int j = Eidx[r];
    sI[tid] = b * cN + i;
    sJ[tid] = b * cN + j;
    sMask[tid] = xmask[b * cN + i] * xmask[b * cN + j];
  }
  __syncthreads();
  // Gather-concat A tile (16 x 384) with float4 loads, f32 -> f16.
  for (int idx = tid; idx < 16 * 96; idx += 256) {
    int row = idx / 96, c4 = idx - row * 96;
    int c = c4 * 4;
    const float* src;
    if (mode == 0) {
      if (c < 128)      src = &hE[(size_t)(r0 + row) * 128 + c];
      else if (c < 256) src = &hV[(size_t)sI[row] * 128 + (c - 128)];
      else              src = &hV[(size_t)sJ[row] * 128 + (c - 256)];
    } else {
      if (c < 128)      src = &hV[(size_t)sI[row] * 128 + c];
      else if (c < 256) src = &hV[(size_t)sJ[row] * 128 + (c - 128)];
      else              src = &hE[(size_t)(r0 + row) * 128 + (c - 256)];
    }
    float4 v = *reinterpret_cast<const float4*>(src);
    _Float16* d = &sA[row * 384 + c];
    d[0] = (_Float16)v.x; d[1] = (_Float16)v.y;
    d[2] = (_Float16)v.z; d[3] = (_Float16)v.w;
  }
  __syncthreads();
  int w = tid >> 5, l = tid & 31;
  int n = w * 16 + (l & 15), mb = (l >> 4) ? 8 : 0;
  // stage 1: 384 -> 128, relu (KT=12)
  v8f acc = {};
#pragma unroll
  for (int kt = 0; kt < 12; ++kt) {
    if (kt + 1 < 12)
      __builtin_prefetch(&W1[(((size_t)w * 12 + kt + 1) * 32 + l) << 4], 0, 1);
    v16h a  = load_a_frag_lds(sA, 384, kt * 32);
    v16h bf = load_b_packed(W1, 12, kt, w);
    acc = wmma_f16(a, bf, acc);
  }
#pragma unroll
  for (int e = 0; e < 8; ++e)
    sM[(mb + e) * 128 + n] = (_Float16)fmaxf(acc[e] + b1[n], 0.f);
  __syncthreads();
  // stage 2: 128 -> 128, relu (KT=4)
  v8f acc2 = {};
#pragma unroll
  for (int kt = 0; kt < 4; ++kt) {
    v16h a  = load_a_frag_lds(sM, 128, kt * 32);
    v16h bf = load_b_packed(W2, 4, kt, w);
    acc2 = wmma_f16(a, bf, acc2);
  }
  __syncthreads();
#pragma unroll
  for (int e = 0; e < 8; ++e)
    sM[(mb + e) * 128 + n] = (_Float16)fmaxf(acc2[e] + b2[n], 0.f);
  __syncthreads();
  // stage 3: 128 -> 128 (KT=4)
  v8f acc3 = {};
#pragma unroll
  for (int kt = 0; kt < 4; ++kt) {
    v16h a  = load_a_frag_lds(sM, 128, kt * 32);
    v16h bf = load_b_packed(W3, 4, kt, w);
    acc3 = wmma_f16(a, bf, acc3);
  }
  if (mode == 1) {
#pragma unroll
    for (int e = 0; e < 8; ++e) {
      int row = mb + e;
      mn_out[(size_t)(r0 + row) * 128 + n] = (acc3[e] + b3[n]) * sMask[row];
    }
    return;
  }
#pragma unroll
  for (int e = 0; e < 8; ++e) {
    int row = mb + e;
    float m3 = (acc3[e] + b3[n]) * sMask[row];
    sT[row * 128 + n] = hE[(size_t)(r0 + row) * 128 + n] + m3;
  }
  __syncthreads();
  ln_tile_16x128(sT, Lg, Lb, rs, rq);
  for (int idx = tid; idx < 16 * 128; idx += 256) {
    int row = idx >> 7, c = idx & 127;
    hE[(size_t)(r0 + row) * 128 + c] = sT[idx];
  }
}

// Aggregate node messages over K neighbors, residual + LN into hV.
__global__ void node_agg_kernel(const float* __restrict__ mn, float* __restrict__ hV,
                                const float* g, const float* be) {
  __shared__ float red[128];
  int node = blockIdx.x, c = threadIdx.x;
  float s = 0.f;
  for (int k = 0; k < cK; ++k) s += mn[((size_t)node * cK + k) * 128 + c];
  float a = hV[(size_t)node * 128 + c] + s * (1.f / (float)cK);
  float mean = block_sum_128(a, red) * (1.f / 128.f);
  float d = a - mean;
  float var = block_sum_128(d * d, red) * (1.f / 128.f);
  hV[(size_t)node * 128 + c] = d * rsqrtf(var + 1e-5f) * g[c] + be[c];
}

// Fused FFN: hV = LN(hV + relu(hV@W1+b1)@W2+b2) * mask
__global__ void ffn_kernel(float* __restrict__ hV, const float* __restrict__ xmask,
                           const _Float16* __restrict__ W1, const float* __restrict__ b1f,
                           const _Float16* __restrict__ W2, const float* __restrict__ b2f,
                           const float* g, const float* be) {
  __shared__ _Float16 sA[16 * 128];
  __shared__ float    sOld[16 * 128];
  __shared__ _Float16 sM1[16 * 512];
  __shared__ float    sT[16 * 128];
  __shared__ float rs[256], rq[256];
  int tid = threadIdx.x;
  int r0 = blockIdx.x * 16;
  for (int idx = tid; idx < 16 * 32; idx += 256) {
    float4 v = *reinterpret_cast<const float4*>(&hV[(size_t)r0 * 128 + idx * 4]);
    sOld[idx * 4 + 0] = v.x; sOld[idx * 4 + 1] = v.y;
    sOld[idx * 4 + 2] = v.z; sOld[idx * 4 + 3] = v.w;
    sA[idx * 4 + 0] = (_Float16)v.x; sA[idx * 4 + 1] = (_Float16)v.y;
    sA[idx * 4 + 2] = (_Float16)v.z; sA[idx * 4 + 3] = (_Float16)v.w;
  }
  __syncthreads();
  int w = tid >> 5, l = tid & 31;
  int n16 = l & 15, mb = (l >> 4) ? 8 : 0;
  // stage 1: 128 -> 512 (KT=4, 32 n-tiles, 4 per wave)
#pragma unroll
  for (int it = 0; it < 4; ++it) {
    int nt = w + 8 * it;
    v8f acc = {};
#pragma unroll
    for (int kt = 0; kt < 4; ++kt) {
      v16h a  = load_a_frag_lds(sA, 128, kt * 32);
      v16h bf = load_b_packed(W1, 4, kt, nt);
      acc = wmma_f16(a, bf, acc);
    }
    int col = nt * 16 + n16;
#pragma unroll
    for (int e = 0; e < 8; ++e)
      sM1[(mb + e) * 512 + col] = (_Float16)fmaxf(acc[e] + b1f[col], 0.f);
  }
  __syncthreads();
  // stage 2: 512 -> 128 (KT=16)
  int n = w * 16 + n16;
  v8f acc2 = {};
#pragma unroll
  for (int kt = 0; kt < 16; ++kt) {
    v16h a  = load_a_frag_lds(sM1, 512, kt * 32);
    v16h bf = load_b_packed(W2, 16, kt, w);
    acc2 = wmma_f16(a, bf, acc2);
  }
#pragma unroll
  for (int e = 0; e < 8; ++e) {
    int row = mb + e;
    sT[row * 128 + n] = sOld[row * 128 + n] + acc2[e] + b2f[n];
  }
  __syncthreads();
  ln_tile_16x128(sT, g, be, rs, rq);
  for (int idx = tid; idx < 16 * 128; idx += 256) {
    int row = idx >> 7, c = idx & 127;
    hV[(size_t)(r0 + row) * 128 + c] = sT[idx] * xmask[r0 + row];
  }
}

// etab = hE @ Woutw + Woutb (400 cols, weights packed+padded to 416)
__global__ void etab_kernel(const float* __restrict__ hE,
                            const _Float16* __restrict__ WoutP,
                            const float* __restrict__ Woutb,
                            float* __restrict__ out) {
  __shared__ _Float16 sA[16 * 128];
  int tid = threadIdx.x;
  int r0 = blockIdx.x * 16;
  for (int idx = tid; idx < 16 * 32; idx += 256) {
    float4 v = *reinterpret_cast<const float4*>(&hE[(size_t)r0 * 128 + idx * 4]);
    sA[idx * 4 + 0] = (_Float16)v.x; sA[idx * 4 + 1] = (_Float16)v.y;
    sA[idx * 4 + 2] = (_Float16)v.z; sA[idx * 4 + 3] = (_Float16)v.w;
  }
  __syncthreads();
  int w = tid >> 5, l = tid & 31;
  int mb = (l >> 4) ? 8 : 0;
  for (int t = w; t < 26; t += 8) {
    v8f acc = {};
#pragma unroll
    for (int kt = 0; kt < 4; ++kt) {
      v16h a  = load_a_frag_lds(sA, 128, kt * 32);
      v16h bf = load_b_packed(WoutP, 4, kt, t);
      acc = wmma_f16(a, bf, acc);
    }
    int n = t * 16 + (l & 15);
    if (n < cOUT) {
#pragma unroll
      for (int e = 0; e < 8; ++e)
        out[(size_t)(r0 + mb + e) * cOUT + n] = acc[e] + Woutb[n];
    }
  }
}

__global__ void sscore_kernel(const float* __restrict__ hV, const float* Sw,
                              const float* Sb, const float* xmask,
                              float* __restrict__ out) {
  int node = blockIdx.x * blockDim.x + threadIdx.x;
  if (node >= NNODE) return;
  float s = Sb[0];
  for (int k = 0; k < 128; ++k) s += hV[(size_t)node * 128 + k] * Sw[k];
  out[node] = s * xmask[node];
}

__global__ void eidx_kernel(const int* __restrict__ Eidx, float* __restrict__ out) {
  int r = blockIdx.x * blockDim.x + threadIdx.x;
  if (r < NEDGE) out[r] = (float)Eidx[r];
}

// Pack f32 weight [Kdim][srcN] -> f16 fragment-major layout [N/16][Kdim/32][32][16],
// element (nt,kt,lane,e) = W[kt*32 + ((lane>=16)?8:0) + (e<8 ? e : e+8)][nt*16 + (lane&15)].
// Columns n >= srcN are zero-padded (used for Woutw 400 -> 416).
__global__ void pack_weight_kernel(const float* __restrict__ src, _Float16* __restrict__ dst,
                                   int Kdim, int N, int srcN) {
  int idx = blockIdx.x * 256 + threadIdx.x;
  int total = Kdim * N;
  if (idx >= total) return;
  int e  = idx & 15;
  int l  = (idx >> 4) & 31;
  int t  = idx >> 9;
  int KT = Kdim >> 5;
  int kt = t % KT, nt = t / KT;
  int k = kt * 32 + ((l >> 4) << 3) + ((e < 8) ? e : (e + 8));
  int n = nt * 16 + (l & 15);
  float v = (n < srcN) ? src[(size_t)k * srcN + n] : 0.f;
  dst[idx] = (_Float16)v;
}

// ---------------------------------------------------------------------------
extern "C" void kernel_launch(void* const* d_in, const int* in_sizes, int n_in,
                              void* d_out, int out_size, void* d_ws, size_t ws_size,
                              hipStream_t stream) {
  (void)in_sizes; (void)n_in; (void)out_size; (void)ws_size;
  const float* X      = (const float*)d_in[0];
  const float* xmask  = (const float*)d_in[1];
  const float* fVw    = (const float*)d_in[3];
  const float* fVb    = (const float*)d_in[4];
  const float* fVg    = (const float*)d_in[5];
  const float* fVbeta = (const float*)d_in[6];
  const float* fEw    = (const float*)d_in[7];
  const float* fEb    = (const float*)d_in[8];
  const float* fEg    = (const float*)d_in[9];
  const float* fEbeta = (const float*)d_in[10];
  const float* Wvw    = (const float*)d_in[11];
  const float* Wvb    = (const float*)d_in[12];
  const float* Wew    = (const float*)d_in[13];
  const float* Web    = (const float*)d_in[14];
  const float* eW1    = (const float*)d_in[15];
  const float* eb1    = (const float*)d_in[16];
  const float* eW2    = (const float*)d_in[17];
  const float* eb2    = (const float*)d_in[18];
  const float* eW3    = (const float*)d_in[19];
  const float* eb3    = (const float*)d_in[20];
  const float* eLg    = (const float*)d_in[21];
  const float* eLb    = (const float*)d_in[22];
  const float* nW1    = (const float*)d_in[23];
  const float* nb1    = (const float*)d_in[24];
  const float* nW2    = (const float*)d_in[25];
  const float* nb2    = (const float*)d_in[26];
  const float* nW3    = (const float*)d_in[27];
  const float* nb3    = (const float*)d_in[28];
  const float* n1g    = (const float*)d_in[29];
  const float* n1b    = (const float*)d_in[30];
  const float* fW1    = (const float*)d_in[31];
  const float* fb1    = (const float*)d_in[32];
  const float* fW2    = (const float*)d_in[33];
  const float* fb2    = (const float*)d_in[34];
  const float* n2g    = (const float*)d_in[35];
  const float* n2b    = (const float*)d_in[36];
  const float* Woutw  = (const float*)d_in[37];
  const float* Woutb  = (const float*)d_in[38];
  const float* Sw     = (const float*)d_in[39];
  const float* Sb     = (const float*)d_in[40];

  char* ws = (char*)d_ws;
  size_t cur = 0;
  auto alloc = [&](size_t bytes) -> void* {
    void* p = ws + cur;
    cur += (bytes + 255) & ~(size_t)255;
    return p;
  };
  float*     hV   = (float*)alloc((size_t)NNODE * cH * 4);
  float*     hE   = (float*)alloc((size_t)NEDGE * cH * 4);
  float*     mn   = (float*)alloc((size_t)NEDGE * cH * 4);
  float*     Vf   = (float*)alloc((size_t)NNODE * 6 * 4);
  float*     Dn   = (float*)alloc((size_t)NEDGE * 4);
  int*       Eix  = (int*)alloc((size_t)NEDGE * 4);
  _Float16*  Ef16 = (_Float16*)alloc((size_t)NEDGE * 32 * 2);
  _Float16* wfEw  = (_Float16*)alloc(32 * 128 * 2);
  _Float16* wWew  = (_Float16*)alloc(128 * 128 * 2);
  _Float16* weW1  = (_Float16*)alloc((size_t)3 * 384 * 128 * 2);
  _Float16* weW2  = (_Float16*)alloc((size_t)3 * 128 * 128 * 2);
  _Float16* weW3  = (_Float16*)alloc((size_t)3 * 128 * 128 * 2);
  _Float16* wnW1  = (_Float16*)alloc((size_t)3 * 384 * 128 * 2);
  _Float16* wnW2  = (_Float16*)alloc((size_t)3 * 128 * 128 * 2);
  _Float16* wnW3  = (_Float16*)alloc((size_t)3 * 128 * 128 * 2);
  _Float16* wfW1  = (_Float16*)alloc((size_t)3 * 128 * 512 * 2);
  _Float16* wfW2  = (_Float16*)alloc((size_t)3 * 512 * 128 * 2);
  _Float16* wWout = (_Float16*)alloc((size_t)128 * 416 * 2);

  auto pack = [&](const float* src, _Float16* dst, int Kdim, int N, int srcN) {
    int total = Kdim * N;
    pack_weight_kernel<<<(total + 255) / 256, 256, 0, stream>>>(src, dst, Kdim, N, srcN);
  };
  pack(fEw, wfEw, 32, 128, 128);
  pack(Wew, wWew, 128, 128, 128);
  for (int l = 0; l < 3; ++l) {
    pack(eW1 + (size_t)l * 384 * 128, weW1 + (size_t)l * 384 * 128, 384, 128, 128);
    pack(eW2 + (size_t)l * 128 * 128, weW2 + (size_t)l * 128 * 128, 128, 128, 128);
    pack(eW3 + (size_t)l * 128 * 128, weW3 + (size_t)l * 128 * 128, 128, 128, 128);
    pack(nW1 + (size_t)l * 384 * 128, wnW1 + (size_t)l * 384 * 128, 384, 128, 128);
    pack(nW2 + (size_t)l * 128 * 128, wnW2 + (size_t)l * 128 * 128, 128, 128, 128);
    pack(nW3 + (size_t)l * 128 * 128, wnW3 + (size_t)l * 128 * 128, 128, 128, 128);
    pack(fW1 + (size_t)l * 128 * 512, wfW1 + (size_t)l * 128 * 512, 128, 512, 512);
    pack(fW2 + (size_t)l * 512 * 128, wfW2 + (size_t)l * 512 * 128, 512, 128, 128);
  }
  pack(Woutw, wWout, 128, 416, 400);

  dihedral_kernel<<<(cB * cN * 3 + 255) / 256, 256, 0, stream>>>(X, Vf);
  topk_kernel<<<NNODE, 256, 0, stream>>>(X, xmask, Eix, Dn);
  ef_kernel<<<(NEDGE + 255) / 256, 256, 0, stream>>>(Dn, Eix, Ef16);
  v_embed_kernel<<<NNODE, 128, 0, stream>>>(Vf, fVw, fVb, fVg, fVbeta, Wvw, Wvb, hV);
  e_embed_kernel<<<NEDGE / 16, 256, 0, stream>>>(Ef16, wfEw, fEb, fEg, fEbeta, wWew, Web, hE);

  for (int l = 0; l < 3; ++l) {
    msg_kernel<<<NEDGE / 16, 256, 0, stream>>>(
        hV, hE, mn, Eix, xmask,
        weW1 + (size_t)l * 384 * 128, eb1 + l * 128,
        weW2 + (size_t)l * 128 * 128, eb2 + l * 128,
        weW3 + (size_t)l * 128 * 128, eb3 + l * 128,
        eLg + l * 128, eLb + l * 128, 0);
    msg_kernel<<<NEDGE / 16, 256, 0, stream>>>(
        hV, hE, mn, Eix, xmask,
        wnW1 + (size_t)l * 384 * 128, nb1 + l * 128,
        wnW2 + (size_t)l * 128 * 128, nb2 + l * 128,
        wnW3 + (size_t)l * 128 * 128, nb3 + l * 128,
        nullptr, nullptr, 1);
    node_agg_kernel<<<NNODE, 128, 0, stream>>>(mn, hV, n1g + l * 128, n1b + l * 128);
    ffn_kernel<<<NNODE / 16, 256, 0, stream>>>(
        hV, xmask,
        wfW1 + (size_t)l * 128 * 512, fb1 + l * 512,
        wfW2 + (size_t)l * 512 * 128, fb2 + l * 128,
        n2g + l * 128, n2b + l * 128);
  }

  float* out      = (float*)d_out;
  float* out_etab = out;
  float* out_eidx = out + (size_t)NEDGE * cOUT;
  float* out_ss   = out_eidx + NEDGE;
  etab_kernel<<<NEDGE / 16, 256, 0, stream>>>(hE, wWout, Woutb, out_etab);
  eidx_kernel<<<(NEDGE + 255) / 256, 256, 0, stream>>>(Eix, out_eidx);
  sscore_kernel<<<(NNODE + 127) / 128, 128, 0, stream>>>(hV, Sw, Sb, xmask, out_ss);
}
